// SAM_85950885527911
// MI455X (gfx1250) — matrix operand order repaired and checked by reading, not compiled, not run on hardware
//
#include <hip/hip_runtime.h>
#include <hip/hip_bf16.h>
#include <math.h>

// Problem constants
#define BATCH 8
#define CH    32
#define HH    256
#define WW    256
#define HW    (HH * WW)

typedef __attribute__((ext_vector_type(16))) __bf16        v16bf;
typedef __attribute__((ext_vector_type(8)))  float         v8f;
typedef __attribute__((ext_vector_type(4)))  unsigned int  v4u;

union CastA { struct { v4u lo, hi; } u; v16bf v; };
union AccF  { v8f v; float f[8]; };

// ---------------------------------------------------------------------------
// Weight repack: OIHW fp32 -> bf16 [tap][j][n][kk]  (tap=dy*3+dx, kk=chan in
// 32-wide K chunk).  This makes each lane's WMMA B-fragment two contiguous
// b128 loads, K contiguous per lane half (same order as the A fragment):
//   lane<16 : N=lane,   K=0..15  -> dwords base+0..7
//   lane>=16: N=lane-16,K=16..31 -> dwords base+8..15
// ---------------------------------------------------------------------------
__global__ __launch_bounds__(256) void repack_w(const float* __restrict__ src,
                                                __bf16* __restrict__ dst,
                                                int nj, int cin, int nel)
{
    int t = blockIdx.x * 256 + threadIdx.x;
    if (t >= nel) return;
    int kk  = t & 31;
    int n   = (t >> 5) & 31;
    int j   = (t >> 10) % nj;
    int tap = t / (1024 * nj);
    int c   = j * 32 + kk;
    int dy  = tap / 3, dx = tap % 3;
    dst[t] = (__bf16)src[((n * cin + c) * 3 + dy) * 3 + dx];
}

// ---------------------------------------------------------------------------
// Implicit-GEMM 3x3 conv (pad=1), CIN -> 32, NCHW fp32 in/out, bf16 WMMA.
// One wave = 16 consecutive-W output pixels x 32 output channels.
// LDS patch: 3 rows x 18 cols x CIN bf16 (channel pairs packed in dwords) so
// the A fragment for a tap/chunk is two ds_load_b128 in contiguous K order,
// as B uses (lane<16: K 0..15; lane>=16: K 16..31).
// ---------------------------------------------------------------------------
template <int CIN>
__global__ __launch_bounds__(128) void conv3x3_wmma(const float* __restrict__ in,
                                                    const __bf16* __restrict__ wgt,
                                                    float* __restrict__ out,
                                                    int do_relu)
{
    constexpr int NJ    = CIN / 32;       // K chunks of 32
    constexpr int PATCH = 3 * 18 * CIN;   // bf16 elements per wave
    __shared__ __align__(16) __bf16 patch[4][PATCH];

    const int wave = threadIdx.x >> 5;
    const int lane = threadIdx.x & 31;
    const int tile = blockIdx.x * 4 + wave;      // B*H*(W/16) = 32768 tiles
    const int w0   = (tile & 15) << 4;
    const int h    = (tile >> 4) & 255;
    const int b    = tile >> 12;

    const float* inb = in + (size_t)b * CIN * HW;

    // ---- stage 3x18xCIN fp32 patch -> LDS bf16 (zero-padded halo) ----------
    {
        const int  col   = lane;
        const bool colok = (col < 18);
        const int  gw    = w0 - 1 + col;
        const bool wok   = colok && (gw >= 0) && (gw < WW);
        for (int r = 0; r < 3; ++r) {
            const int  gh = h - 1 + r;
            const bool ok = wok && (gh >= 0) && (gh < HH);
            long off = (long)gh * WW + gw;
            const float* s = inb + (ok ? off : 0);
            __bf16* d = &patch[wave][(r * 18 + col) * CIN];
            if (colok) {
                if (ok) __builtin_prefetch(s + 18, 0, 0);   // next tile's row
                #pragma unroll 8
                for (int c = 0; c < CIN; c += 2) {
                    float f0 = 0.0f, f1 = 0.0f;
                    if (ok) { f0 = s[0]; f1 = s[HW]; }
                    d[0] = (__bf16)f0;
                    d[1] = (__bf16)f1;
                    s += 2 * (size_t)HW;
                    d += 2;
                }
            }
        }
    }
    __threadfence_block();   // order LDS stores vs cross-lane fragment loads

    const int m  = lane & 15;   // output pixel row of A / column of B,C
    const int kh = lane >> 4;   // K-half select

    v8f acc0 = {};
    v8f acc1 = {};

    const unsigned int* lp = (const unsigned int*)&patch[wave][0];
    const unsigned int* wp = (const unsigned int*)wgt;

    for (int tap = 0; tap < 9; ++tap) {
        const int r   = tap / 3;
        const int dxi = tap % 3;
        const int pix = r * 18 + m + dxi;     // patch pixel for this tap
        #pragma unroll
        for (int j = 0; j < NJ; ++j) {
            // A fragment: dwords [base..base+3] and [base+4..base+7]
            const int abase = pix * (CIN / 2) + j * 16 + kh * 8;
            CastA af;
            af.u.lo = *(const v4u*)(lp + abase);
            af.u.hi = *(const v4u*)(lp + abase + 4);
            // B fragments (two N tiles): 8 contiguous dwords each
            const int wb0 = ((tap * NJ + j) * 32 + m) * 16 + kh * 8;
            CastA bf0, bf1;
            bf0.u.lo = *(const v4u*)(wp + wb0);
            bf0.u.hi = *(const v4u*)(wp + wb0 + 4);
            bf1.u.lo = *(const v4u*)(wp + wb0 + 256);   // n += 16 -> +16*16 dwords
            bf1.u.hi = *(const v4u*)(wp + wb0 + 260);
            acc0 = __builtin_amdgcn_wmma_f32_16x16x32_bf16(
                       false, af.v, false, bf0.v, (short)0, acc0, false, false);
            acc1 = __builtin_amdgcn_wmma_f32_16x16x32_bf16(
                       false, af.v, false, bf1.v, (short)0, acc1, false, false);
        }
    }

    // ---- LDS bounce (pitch 17) so global stores are coalesced along W ------
    __threadfence_block();
    float* cb = (float*)&patch[wave][0];    // reuse patch (>= 32*17 floats)
    AccF a0, a1; a0.v = acc0; a1.v = acc1;
    #pragma unroll
    for (int i = 0; i < 8; ++i) {
        // C/D layout: lane L -> N = L&15, M = vgpr + 8*(L>>4)
        cb[m * 17 + kh * 8 + i]        = a0.f[i];
        cb[(m + 16) * 17 + kh * 8 + i] = a1.f[i];
    }
    __threadfence_block();

    float* ob = out + (size_t)b * 32 * HW + (size_t)h * WW + w0;
    #pragma unroll
    for (int k = 0; k < 16; ++k) {
        const int ch = 2 * k + kh;
        float v = cb[ch * 17 + m];
        if (do_relu) v = fmaxf(v, 0.0f);
        ob[(size_t)ch * HW + m] = v;
    }
}

// ---------------------------------------------------------------------------
// Vertical IRNN scans (up dir=0 -> z[:,0:32], down dir=2 -> z[:,64:96]).
// Thread per (b,c,w) column; loads/stores coalesced along W at every step.
// Semantics: first element passes through, then h = relu(x + w*h + b).
// ---------------------------------------------------------------------------
__global__ __launch_bounds__(256) void scan_v(const float* __restrict__ in,
                                              float* __restrict__ z,
                                              const float* __restrict__ wr,
                                              const float* __restrict__ br)
{
    const int  bc  = blockIdx.x;
    const int  b   = bc >> 5;
    const int  c   = bc & 31;
    const int  w   = threadIdx.x;
    const bool up  = (blockIdx.y == 1);
    const int  dir = up ? 0 : 2;
    const float wv = wr[dir * CH + c];
    const float bv = br[dir * CH + c];
    const float* ib = in + (size_t)bc * HW;
    float* zb = z + ((size_t)b * 4 * CH + dir * CH + c) * HW;

    if (!up) {
        float y = ib[w];
        zb[w] = y;
        for (int h = 1; h < HH; ++h) {
            float v = ib[h * WW + w];
            y = fmaxf(v + wv * y + bv, 0.0f);
            zb[h * WW + w] = y;
        }
    } else {
        float y = ib[(HH - 1) * WW + w];
        zb[(HH - 1) * WW + w] = y;
        for (int h = HH - 2; h >= 0; --h) {
            float v = ib[h * WW + w];
            y = fmaxf(v + wv * y + bv, 0.0f);
            zb[h * WW + w] = y;
        }
    }
}

// ---------------------------------------------------------------------------
// Horizontal IRNN scans (right dir=1 -> z[:,32:64], left dir=3 -> z[:,96:128]).
// Block owns one (b,c) plane; 32-wide LDS transpose tiles (pitch 33,
// conflict-free) keep both global phases coalesced; in-place scan per row.
// ---------------------------------------------------------------------------
__global__ __launch_bounds__(256) void scan_h(const float* __restrict__ in,
                                              float* __restrict__ z,
                                              const float* __restrict__ wr,
                                              const float* __restrict__ br)
{
    __shared__ float tile[256][33];
    const int  bc  = blockIdx.x;
    const int  b   = bc >> 5;
    const int  c   = bc & 31;
    const bool rev = (blockIdx.y == 1);
    const int  dir = rev ? 3 : 1;
    const int  t   = threadIdx.x;
    const float wv = wr[dir * CH + c];
    const float bv = br[dir * CH + c];
    const float* ib = in + (size_t)bc * HW;
    float* zb = z + ((size_t)b * 4 * CH + dir * CH + c) * HW;

    const int rr  = t >> 5;
    const int col = t & 31;
    float y = 0.0f;

    for (int cc = 0; cc < 8; ++cc) {
        const int chunk = rev ? 7 - cc : cc;
        const int w0    = chunk << 5;
        __syncthreads();
        for (int k = 0; k < 32; ++k) {
            const int row = (k << 3) + rr;
            tile[row][col] = ib[row * WW + w0 + col];
        }
        __syncthreads();
        if (!rev) {
            for (int j = 0; j < 32; ++j) {
                float v = tile[t][j];
                y = (w0 + j == 0) ? v : fmaxf(v + wv * y + bv, 0.0f);
                tile[t][j] = y;
            }
        } else {
            for (int j = 31; j >= 0; --j) {
                float v = tile[t][j];
                y = (w0 + j == WW - 1) ? v : fmaxf(v + wv * y + bv, 0.0f);
                tile[t][j] = y;
            }
        }
        __syncthreads();
        for (int k = 0; k < 32; ++k) {
            const int row = (k << 3) + rr;
            zb[row * WW + w0 + col] = tile[row][col];
        }
    }
}

// ---------------------------------------------------------------------------
// 1x1 conv (32 -> 1) + sigmoid epilogue.
// ---------------------------------------------------------------------------
__global__ __launch_bounds__(256) void conv1x1_sigmoid(const float* __restrict__ y,
                                                       const float* __restrict__ wo,
                                                       float* __restrict__ out)
{
    const int t = blockIdx.x * 256 + threadIdx.x;   // B*H*W = 524288 threads
    const int b = t >> 16;
    const int p = t & 65535;
    const float* yb = y + (size_t)b * CH * HW + p;
    float s = 0.0f;
    #pragma unroll
    for (int c = 0; c < CH; ++c) s += yb[(size_t)c * HW] * wo[c];
    out[t] = 1.0f / (1.0f + __expf(-s));
}

// ---------------------------------------------------------------------------
extern "C" void kernel_launch(void* const* d_in, const int* in_sizes, int n_in,
                              void* d_out, int out_size, void* d_ws, size_t ws_size,
                              hipStream_t stream)
{
    (void)in_sizes; (void)n_in; (void)out_size; (void)ws_size;

    const float* x          = (const float*)d_in[0];
    const float* conv_in_w  = (const float*)d_in[1];
    const float* conv2_w    = (const float*)d_in[2];
    const float* conv3_w    = (const float*)d_in[3];
    const float* conv_out_w = (const float*)d_in[4];
    const float* irnn1_w    = (const float*)d_in[5];
    const float* irnn1_b    = (const float*)d_in[6];
    const float* irnn2_w    = (const float*)d_in[7];
    const float* irnn2_b    = (const float*)d_in[8];
    float* out = (float*)d_out;

    // Workspace layout (bytes):
    //   yA   @ 0          : 64 MiB   (B,32,H,W) fp32
    //   z    @ 64 MiB     : 256 MiB  (B,128,H,W) fp32 (reused for both irnns)
    //   yB   @ 320 MiB    : 64 MiB
    //   wbin @ 384 MiB    : 18 KiB   repacked bf16 weights
    //   wb2  @ +18432     : 72 KiB
    //   wb3  @ +73728     : 72 KiB
    char* ws = (char*)d_ws;
    float*  yA   = (float*)(ws);
    float*  z    = (float*)(ws + 67108864ull);
    float*  yB   = (float*)(ws + 335544320ull);
    __bf16* wbin = (__bf16*)(ws + 402653184ull);
    __bf16* wb2  = (__bf16*)(ws + 402671616ull);
    __bf16* wb3  = (__bf16*)(ws + 402745344ull);

    // weight repacks (tiny)
    repack_w<<<36,  256, 0, stream>>>(conv_in_w, wbin, 1, 32,  9216);
    repack_w<<<144, 256, 0, stream>>>(conv2_w,   wb2,  4, 128, 36864);
    repack_w<<<144, 256, 0, stream>>>(conv3_w,   wb3,  4, 128, 36864);

    const dim3 gscan(BATCH * CH, 2);

    // conv_in: x -> yA
    conv3x3_wmma<32><<<8192, 128, 0, stream>>>(x, wbin, yA, 0);

    // irnn1: yA -> z (up/down via scan_v, right/left via scan_h)
    scan_v<<<gscan, 256, 0, stream>>>(yA, z, irnn1_w, irnn1_b);
    scan_h<<<gscan, 256, 0, stream>>>(yA, z, irnn1_w, irnn1_b);

    // conv2: z -> yB
    conv3x3_wmma<128><<<8192, 128, 0, stream>>>(z, wb2, yB, 0);

    // irnn2: yB -> z
    scan_v<<<gscan, 256, 0, stream>>>(yB, z, irnn2_w, irnn2_b);
    scan_h<<<gscan, 256, 0, stream>>>(yB, z, irnn2_w, irnn2_b);

    // conv3 (+relu): z -> yA
    conv3x3_wmma<128><<<8192, 128, 0, stream>>>(z, wb3, yA, 1);

    // conv_out + sigmoid: yA -> out
    conv1x1_sigmoid<<<2048, 256, 0, stream>>>(yA, conv_out_w, out);
}
